// ResidualGATLayer_88527865905192
// MI455X (gfx1250) — matrix operand order repaired and checked by reading, not compiled, not run on hardware
//
#include <hip/hip_runtime.h>
#include <hip/hip_bf16.h>

// Residual edge-gated graph-transformer layer for MI455X (gfx1250).
// B=2, V=384, H=128, NH=8, HD=16.
// Heavy GEMM (e@Wu, 9.66 GFLOP) runs on v_wmma_f32_16x16x32_bf16 with the
// Wu^T tile DMA'd into LDS by the Tensor Data Mover; attention GEMMs run on
// v_wmma_f32_16x16x4_f32. Streaming passes over the 151MB edge tensor use
// non-temporal cache hints so the e_tmp intermediate stays resident in the
// 192MB L2 between pass 1 and pass 2.

#if defined(__HIP_DEVICE_COMPILE__) && __has_builtin(__builtin_amdgcn_tensor_load_to_lds) && __has_builtin(__builtin_amdgcn_s_wait_tensorcnt)
#define USE_TDM 1
#endif

namespace {
constexpr int  Bq  = 2;
constexpr int  Vq  = 384;
constexpr long MEq = (long)Bq * Vq * Vq;   // 294912 edge rows

typedef __attribute__((ext_vector_type(16))) __bf16 v16bf;
typedef __attribute__((ext_vector_type(8)))  __bf16 v8bf;
typedef __attribute__((ext_vector_type(8)))  float  v8f;
typedef __attribute__((ext_vector_type(4)))  float  v4f;
typedef __attribute__((ext_vector_type(2)))  float  v2f;
typedef __attribute__((ext_vector_type(4)))  unsigned int v4u;
typedef __attribute__((ext_vector_type(8)))  int    v8i;
typedef __attribute__((ext_vector_type(4)))  int    v4i;

// ---------- wave32 WMMA fragment helpers (CDNA5 ISA 7.12.2 layouts) ----------

// bf16 A-frag (16x32) from row-major bf16 source.
__device__ __forceinline__ v16bf frag_a_bf16(const __bf16* A, int lda, long m0, int k0, int lane) {
  long m = m0 + (lane & 15);
  int  hh = (lane >> 4) & 1;
  const __bf16* p = A + m * lda + k0 + hh * 8;
  v8bf lo = *(const v8bf*)(p);
  v8bf hi = *(const v8bf*)(p + 16);
  v16bf r;
#pragma unroll
  for (int i = 0; i < 8; ++i) { r[i] = lo[i]; r[i + 8] = hi[i]; }
  return r;
}

// bf16 A-frag (16x32) converted on the fly from an f32 source, loaded with
// non-temporal hint (streaming read of the 151MB e tensor - keep L2 clean).
__device__ __forceinline__ v16bf frag_a_f32src_nt(const float* A, int lda, long m0, int k0, int lane) {
  long m = m0 + (lane & 15);
  int  hh = (lane >> 4) & 1;
  const float* p = A + m * lda + k0 + hh * 8;
  v4f c0 = __builtin_nontemporal_load((const v4f*)(p));
  v4f c1 = __builtin_nontemporal_load((const v4f*)(p + 4));
  v4f c2 = __builtin_nontemporal_load((const v4f*)(p + 16));
  v4f c3 = __builtin_nontemporal_load((const v4f*)(p + 20));
  v16bf r;
#pragma unroll
  for (int i = 0; i < 4; ++i) {
    r[i]      = (__bf16)c0[i];
    r[i + 4]  = (__bf16)c1[i];
    r[i + 8]  = (__bf16)c2[i];
    r[i + 12] = (__bf16)c3[i];
  }
  return r;
}

// bf16 B-frag (32x16) from N-major (pre-transposed) weights WT[n][k].
// Works for both global and LDS source pointers (addrspace inferred).
__device__ __forceinline__ v16bf frag_b_bf16(const __bf16* WT, int ldk, int n0, int k0, int lane) {
  int n = n0 + (lane & 15);
  int hh = (lane >> 4) & 1;
  const __bf16* p = WT + (long)n * ldk + k0 + hh * 8;
  v8bf lo = *(const v8bf*)(p);
  v8bf hi = *(const v8bf*)(p + 16);
  v16bf r;
#pragma unroll
  for (int i = 0; i < 8; ++i) { r[i] = lo[i]; r[i + 8] = hi[i]; }
  return r;
}

// f32 A-frag (16x4), row-major source.
__device__ __forceinline__ v2f frag_a_f32(const float* A, int lda, long m0, int k0, int lane) {
  long m = m0 + (lane & 15);
  int hh = lane >> 4;
  const float* p = A + m * lda + k0 + hh * 2;
  v2f r; r[0] = p[0]; r[1] = p[1];
  return r;
}

// f32 B-frag (4x16) from row-major B[k][n].
__device__ __forceinline__ v2f frag_b_f32(const float* Bm, int ldb, int k0, int n0, int lane) {
  int n = n0 + (lane & 15);
  int hh = lane >> 4;
  const float* p = Bm + (long)(k0 + hh * 2) * ldb + n;
  v2f r; r[0] = p[0]; r[1] = p[ldb];
  return r;
}

// f32 B-frag (4x16) from N-major source Bt[n][k] (i.e. B = Bt^T) -> used for Q@K^T.
__device__ __forceinline__ v2f frag_bt_f32(const float* Bt, int ldk, int k0, int n0, int lane) {
  int n = n0 + (lane & 15);
  int hh = lane >> 4;
  const float* p = Bt + (long)n * ldk + k0 + hh * 2;
  v2f r; r[0] = p[0]; r[1] = p[1];
  return r;
}

// Store f32 16x16 C/D frag: VGPR r holds row (m0 + half*8 + r), col = lane&15.
__device__ __forceinline__ void store_c(float* C, int ldc, long m0, int n0, int lane, v8f acc) {
  int n = n0 + (lane & 15);
  long mb = m0 + ((lane >> 4) & 1) * 8;
#pragma unroll
  for (int r = 0; r < 8; ++r) C[(mb + r) * ldc + n] = acc[r];
}
} // namespace

// ---------------- K0: weight transpose+bf16 convert, x->bf16, zero BN sums ----------------
__global__ void k0_prep(const float* Wu, const float* Wv, const float* Wq,
                        const float* Wk, const float* Wval, const float* x,
                        __bf16* WT, __bf16* xb, float* sums) {
  long idx = (long)blockIdx.x * blockDim.x + threadIdx.x;
  if (idx < 5 * 16384) {
    int w = (int)(idx >> 14);
    int r = (int)(idx & 16383);
    int n = r >> 7, k = r & 127;
    const float* W = (w == 0) ? Wu : (w == 1) ? Wv : (w == 2) ? Wq : (w == 3) ? Wk : Wval;
    WT[idx] = (__bf16)W[k * 128 + n];   // WT[w][n][k] = W[k][n]
  } else if (idx < 5 * 16384 + 98304) {
    long i = idx - 5 * 16384;
    xb[i] = (__bf16)x[i];
  } else if (idx < 5 * 16384 + 98304 + 512) {
    sums[idx - (5 * 16384 + 98304)] = 0.0f;
  }
}

// ---------------- K1: Vx/Q/K/Vatt = x@W + b (bf16 WMMA) ----------------
__global__ void k1_node_linear(const __bf16* xb, const __bf16* WT,
                               const float* bv, const float* bq, const float* bk,
                               const float* bval, float* out) {
  int mt = blockIdx.x, nt = blockIdx.y, w = blockIdx.z;
  int lane = threadIdx.x & 31;
  const __bf16* WTw = WT + (long)(1 + w) * 16384;          // slot1..4 = Wv,Wq,Wk,Wval
  const float* bias = (w == 0) ? bv : (w == 1) ? bq : (w == 2) ? bk : bval;
  v8f acc = {};
#pragma unroll
  for (int kt = 0; kt < 4; ++kt) {
    v16bf a = frag_a_bf16(xb, 128, (long)mt * 16, kt * 32, lane);
    v16bf b = frag_b_bf16(WTw, 128, nt * 16, kt * 32, lane);
    acc = __builtin_amdgcn_wmma_f32_16x16x32_bf16(false, a, false, b, (short)0, acc, false, false);
  }
  int n = nt * 16 + (lane & 15);
  float bs = bias[n];
#pragma unroll
  for (int r = 0; r < 8; ++r) acc[r] += bs;
  store_c(out + (long)w * 98304, 128, (long)mt * 16, nt * 16, lane, acc);
}

// ---------------- K2: e_tmp = e@Wu + bu + Vx_i + Vx_j; gate mean; BN sums ----------------
// Wu^T (bf16, N-major, 32KB) is DMA'd into LDS once per workgroup by the TDM,
// then all 8 waves feed B-fragments from LDS (ds_load) while VMEM streams e.
__global__ void k2_edge_gemm(const float* __restrict__ e, const __bf16* __restrict__ WuT,
                             const float* __restrict__ bu, const float* __restrict__ Vx,
                             float* __restrict__ e_tmp, float* __restrict__ edge_mask,
                             float* __restrict__ edge_sums) {
  __shared__ __bf16 WuLDS[128 * 128];           // 32 KB
  __shared__ float lsum[128], lsq[128];
  if (threadIdx.x < 128) { lsum[threadIdx.x] = 0.f; lsq[threadIdx.x] = 0.f; }

#if defined(USE_TDM)
  if (threadIdx.x < 32) {                       // one wave issues the tensor DMA
    unsigned lds_off = (unsigned)(size_t)(void*)WuLDS;
    unsigned long long ga = (unsigned long long)(size_t)WuT;
    // D# group 0: count=1, lds_addr, 57-bit global addr, type=2 ("image")
    v4u g0;
    g0[0] = 1u;
    g0[1] = lds_off;
    g0[2] = (unsigned)(ga & 0xFFFFFFFFu);
    g0[3] = (unsigned)((ga >> 32) & 0x1FFFFFFu) | (2u << 30);
    // D# group 1: data_size=2B, tensor 128x128, tile 128x128, stride0=128
    v8i g1;
    g1[0] = (1 << 16);        // data_size code 1 (2 bytes), no multicast
    g1[1] = (128 << 16);      // tensor_dim0[15:0]=128 (bits 63:48)
    g1[2] = (128 << 16);      // tensor_dim1[15:0]=128 (bits 95:80)
    g1[3] = (128 << 16);      // tile_dim0=128 (bits 127:112)
    g1[4] = 128;              // tile_dim1=128 (bits 143:128), tile_dim2=0
    g1[5] = 128;              // tensor_dim0_stride[31:0]=128 elements
    g1[6] = 0;
    g1[7] = 0;
    v4i g2 = {0, 0, 0, 0};
    v4i g3 = {0, 0, 0, 0};
#if defined(__clang_major__) && __clang_major__ >= 23
    v8i g1b = {0, 0, 0, 0, 0, 0, 0, 0};
    __builtin_amdgcn_tensor_load_to_lds(g0, g1, g2, g3, g1b, 0);
#else
    __builtin_amdgcn_tensor_load_to_lds(g0, g1, g2, g3, 0);
#endif
    __builtin_amdgcn_s_wait_tensorcnt(0);       // s_wait_tensorcnt 0
  }
#else
  for (int i = threadIdx.x; i < (128 * 128) / 8; i += 256)
    ((v8bf*)WuLDS)[i] = ((const v8bf*)WuT)[i];
#endif
  __syncthreads();

  int wave = threadIdx.x >> 5;
  int lane = threadIdx.x & 31;
  long mtile = (long)blockIdx.x * 8 + wave;
  long m0 = mtile * 16;                 // 384 % 16 == 0 -> tile has fixed (b,i), 16 consecutive j

  int bi = (int)(m0 / Vq);              // b*V + i
  int j0 = (int)(m0 - (long)bi * Vq);
  int bb = bi / Vq;                     // batch
  int hh = (lane >> 4) & 1;
  int jrow = bb * Vq + j0 + hh * 8;     // base (b*V + j) for this half's 8 C rows
  long mrow = m0 + hh * 8;

  v16bf a[4];
#pragma unroll
  for (int kt = 0; kt < 4; ++kt) a[kt] = frag_a_f32src_nt(e, 128, m0, kt * 32, lane);

  float gacc[8] = {0.f, 0.f, 0.f, 0.f, 0.f, 0.f, 0.f, 0.f};

  for (int nt = 0; nt < 8; ++nt) {
    v8f acc = {};
#pragma unroll
    for (int kt = 0; kt < 4; ++kt) {
      v16bf b = frag_b_bf16(WuLDS, 128, nt * 16, kt * 32, lane);
      acc = __builtin_amdgcn_wmma_f32_16x16x32_bf16(false, a[kt], false, b, (short)0, acc, false, false);
    }
    int n = nt * 16 + (lane & 15);
    float base = bu[n] + Vx[bi * 128 + n];
    float s = 0.f, s2 = 0.f;
#pragma unroll
    for (int r = 0; r < 8; ++r) {
      float t = acc[r] + base + Vx[(jrow + r) * 128 + n];
      e_tmp[(mrow + r) * 128 + n] = t;          // RT store: keep e_tmp L2-resident for pass 2
      gacc[r] += 1.f / (1.f + __expf(-t));      // sigmoid gate
      s += t; s2 += t * t;
    }
    s  += __shfl_xor(s, 16);                    // both halves hold same channel
    s2 += __shfl_xor(s2, 16);
    if (lane < 16) { atomicAdd(&lsum[n], s); atomicAdd(&lsq[n], s2); }
  }

  // gate mean over 128 channels (16 lanes/half x 8 n-tiles already summed in gacc)
#pragma unroll
  for (int r = 0; r < 8; ++r) {
    float g = gacc[r];
    g += __shfl_xor(g, 1);
    g += __shfl_xor(g, 2);
    g += __shfl_xor(g, 4);
    g += __shfl_xor(g, 8);
    if ((lane & 15) == 0) edge_mask[mrow + r] = g * (1.0f / 128.0f);
  }

  __syncthreads();
  if (threadIdx.x < 128) {
    atomicAdd(&edge_sums[threadIdx.x], lsum[threadIdx.x]);
    atomicAdd(&edge_sums[128 + threadIdx.x], lsq[threadIdx.x]);
  }
}

// ---------------- K3: scores = Q@K^T per (b,h) (f32 WMMA, K=16) ----------------
__global__ void k3_scores(const float* Qf, const float* Kf, float* scores) {
  int lane = threadIdx.x & 31;
  int mt = blockIdx.x, nt = blockIdx.y, bh = blockIdx.z;
  int b = bh >> 3, h = bh & 7;
  const float* Qp = Qf + (long)b * Vq * 128 + h * 16;
  const float* Kp = Kf + (long)b * Vq * 128 + h * 16;
  v8f acc = {};
#pragma unroll
  for (int kt = 0; kt < 4; ++kt) {
    v2f a = frag_a_f32(Qp, 128, (long)mt * 16, kt * 4, lane);
    v2f bfr = frag_bt_f32(Kp, 128, kt * 4, nt * 16, lane);
    acc = __builtin_amdgcn_wmma_f32_16x16x4_f32(false, a, false, bfr, (short)0, acc, false, false);
  }
  store_c(scores + (long)bh * Vq * Vq, Vq, (long)mt * 16, nt * 16, lane, acc);
}

// ---------------- K4: attn = softmax(scores * mask / sqrt(HD)) row-wise ----------------
__global__ void k4_softmax(float* scores, const float* edge_mask) {
  int row = blockIdx.x;                 // bh*V + i
  int bh = row / Vq, i = row - bh * Vq;
  int b = bh >> 3;
  float* s = scores + (long)row * Vq;
  const float* mrow = edge_mask + ((long)b * Vq + i) * Vq;
  int t = threadIdx.x;
  float v[3];
  float lmax = -3.4e38f;
#pragma unroll
  for (int c = 0; c < 3; ++c) {
    int j = t + c * 128;
    v[c] = s[j] * mrow[j] * 0.25f;      // 1/sqrt(16) = 0.25
    lmax = fmaxf(lmax, v[c]);
  }
  __shared__ float red[128];
  red[t] = lmax; __syncthreads();
  for (int o = 64; o > 0; o >>= 1) { if (t < o) red[t] = fmaxf(red[t], red[t + o]); __syncthreads(); }
  float rmax = red[0];
  __syncthreads();
  float lsum = 0.f;
#pragma unroll
  for (int c = 0; c < 3; ++c) { v[c] = __expf(v[c] - rmax); lsum += v[c]; }
  red[t] = lsum; __syncthreads();
  for (int o = 64; o > 0; o >>= 1) { if (t < o) red[t] += red[t + o]; __syncthreads(); }
  float inv = 1.0f / red[0];
#pragma unroll
  for (int c = 0; c < 3; ++c) s[t + c * 128] = v[c] * inv;
}

// ---------------- K5: attended = attn @ Vatt per (b,h) (f32 WMMA, K=384) ----------------
__global__ void k5_attend(const float* attn, const float* Vattf, float* attended) {
  int lane = threadIdx.x & 31;
  int mt = blockIdx.x, bh = blockIdx.y;
  int b = bh >> 3, h = bh & 7;
  const float* Ap = attn + (long)bh * Vq * Vq;
  const float* Vp = Vattf + (long)b * Vq * 128 + h * 16;
  v8f acc = {};
  for (int kt = 0; kt < 96; ++kt) {
    v2f a = frag_a_f32(Ap, Vq, (long)mt * 16, kt * 4, lane);
    v2f bfr = frag_b_f32(Vp, 128, kt * 4, 0, lane);
    acc = __builtin_amdgcn_wmma_f32_16x16x4_f32(false, a, false, bfr, (short)0, acc, false, false);
  }
  // stores directly in transpose(0,2,1,3).reshape(B,V,H) layout
  store_c(attended + (long)b * Vq * 128 + h * 16, 128, (long)mt * 16, 0, lane, acc);
}

// ---------------- K6: x_tmp = attended @ Wo + bo ; node BN sums ----------------
__global__ void k6_xproj(const float* attended, const float* Wo, const float* bo,
                         float* x_tmp, float* node_sums) {
  int nt = threadIdx.x >> 5;
  int lane = threadIdx.x & 31;
  int mt = blockIdx.x;
  v8f acc = {};
  for (int kt = 0; kt < 32; ++kt) {
    v2f a = frag_a_f32(attended, 128, (long)mt * 16, kt * 4, lane);
    v2f bfr = frag_b_f32(Wo, 128, kt * 4, nt * 16, lane);
    acc = __builtin_amdgcn_wmma_f32_16x16x4_f32(false, a, false, bfr, (short)0, acc, false, false);
  }
  int n = nt * 16 + (lane & 15);
  float bs = bo[n];
  float s = 0.f, s2 = 0.f;
  long mb = (long)mt * 16 + ((lane >> 4) & 1) * 8;
#pragma unroll
  for (int r = 0; r < 8; ++r) {
    float tv = acc[r] + bs;
    x_tmp[(mb + r) * 128 + n] = tv;
    s += tv; s2 += tv * tv;
  }
  s  += __shfl_xor(s, 16);
  s2 += __shfl_xor(s2, 16);
  if (lane < 16) { atomicAdd(&node_sums[n], s); atomicAdd(&node_sums[128 + n], s2); }
}

// ---------------- K7: finalize BN statistics ----------------
__global__ void k7_stats(const float* sums, float* stats) {
  int c = threadIdx.x;
  float invE = 1.0f / (float)MEq;
  float invX = 1.0f / 768.0f;
  float em = sums[c] * invE;
  float ev = sums[128 + c] * invE - em * em;
  float xm = sums[256 + c] * invX;
  float xv = sums[384 + c] * invX - xm * xm;
  stats[c]       = em;
  stats[128 + c] = rsqrtf(ev + 1e-5f);
  stats[256 + c] = xm;
  stats[384 + c] = rsqrtf(xv + 1e-5f);
}

// ---------------- K8: e_new = e + relu(BN(e_tmp)) (in place on d_out) ----------------
// e_tmp load = regular-temporal (L2 hits from K2's stores); e read + e_new
// write = non-temporal (pure streaming, never re-referenced).
__global__ void k8_edge_out(const float* __restrict__ e_in, float* __restrict__ e_io,
                            const float* __restrict__ stats,
                            const float* __restrict__ ew, const float* __restrict__ eb) {
  long idx = (long)blockIdx.x * blockDim.x + threadIdx.x;   // float4 index
  long base = idx * 4;
  int n = (int)(base & 127);
  __builtin_prefetch(e_io + base + 16384, 0, 1);            // global_prefetch_b8 on the RT stream
  v4f t  = *(const v4f*)(e_io + base);
  v4f ei = __builtin_nontemporal_load((const v4f*)(e_in + base));
  v4f o;
#pragma unroll
  for (int c = 0; c < 4; ++c) {
    float r = (t[c] - stats[n + c]) * stats[128 + n + c] * ew[n + c] + eb[n + c];
    o[c] = ei[c] + fmaxf(r, 0.f);
  }
  __builtin_nontemporal_store(o, (v4f*)(e_io + base));
}

// ---------------- K9: x_new = x + relu(BN(x_tmp)) (in place on d_out) ----------------
__global__ void k9_node_out(const float* __restrict__ x_in, float* __restrict__ x_io,
                            const float* __restrict__ stats,
                            const float* __restrict__ nw, const float* __restrict__ nb) {
  long idx = (long)blockIdx.x * blockDim.x + threadIdx.x;   // float4 index, 24576 total
  long base = idx * 4;
  int n = (int)(base & 127);
  v4f t  = *(const v4f*)(x_io + base);
  v4f xi = *(const v4f*)(x_in + base);
  v4f o;
#pragma unroll
  for (int c = 0; c < 4; ++c) {
    float r = (t[c] - stats[256 + n + c]) * stats[384 + n + c] * nw[n + c] + nb[n + c];
    o[c] = xi[c] + fmaxf(r, 0.f);
  }
  *(v4f*)(x_io + base) = o;
}

extern "C" void kernel_launch(void* const* d_in, const int* in_sizes, int n_in,
                              void* d_out, int out_size, void* d_ws, size_t ws_size,
                              hipStream_t stream) {
  (void)in_sizes; (void)n_in; (void)out_size; (void)ws_size;
  const float* x    = (const float*)d_in[0];
  const float* e    = (const float*)d_in[1];
  const float* Wu   = (const float*)d_in[2];
  const float* bu   = (const float*)d_in[3];
  const float* Wv   = (const float*)d_in[4];
  const float* bv   = (const float*)d_in[5];
  const float* Wq   = (const float*)d_in[6];
  const float* bq   = (const float*)d_in[7];
  const float* Wk   = (const float*)d_in[8];
  const float* bk   = (const float*)d_in[9];
  const float* Wval = (const float*)d_in[10];
  const float* bval = (const float*)d_in[11];
  const float* Wo   = (const float*)d_in[12];
  const float* bo   = (const float*)d_in[13];
  const float* bnw  = (const float*)d_in[14];
  const float* bnb  = (const float*)d_in[15];
  const float* bew  = (const float*)d_in[16];
  const float* beb  = (const float*)d_in[17];

  float* outX = (float*)d_out;                 // x_new [768,128]
  float* outE = (float*)d_out + 98304;         // e_new [294912,128]; holds e_tmp between K2 and K8

  char* ws = (char*)d_ws;
  __bf16* WT    = (__bf16*)(ws);               // 5 x [128][128] bf16: Wu,Wv,Wq,Wk,Wval (N-major)
  __bf16* xb    = (__bf16*)(ws + 163840);      // x as bf16
  float*  Vx    = (float*)(ws + 360448);       // Vx,Q,K,Vatt consecutive fp32 [768][128]
  float*  Qf    = (float*)(ws + 753664);
  float*  Kf    = (float*)(ws + 1146880);
  float*  Vattf = (float*)(ws + 1540096);
  float*  attd  = (float*)(ws + 1933312);      // attended [768][128]
  float*  emask = (float*)(ws + 2326528);      // edge_mask [294912]
  float*  sums  = (float*)(ws + 3506176);      // 512 f32: edge sum/sq | node sum/sq
  float*  stats = (float*)(ws + 3508224);      // 512 f32: em,erstd,xm,xrstd
  float*  scores= (float*)(ws + 3510272);      // [16][384][384] fp32 (attn in place)

  k0_prep<<<707, 256, 0, stream>>>(Wu, Wv, Wq, Wk, Wval, x, WT, xb, sums);
  k1_node_linear<<<dim3(48, 8, 4), 32, 0, stream>>>(xb, WT, bv, bq, bk, bval, Vx);
  k2_edge_gemm<<<2304, 256, 0, stream>>>(e, WT, bu, Vx, outE, emask, sums);
  k3_scores<<<dim3(24, 24, 16), 32, 0, stream>>>(Qf, Kf, scores);
  k4_softmax<<<6144, 128, 0, stream>>>(scores, emask);
  k5_attend<<<dim3(24, 16), 32, 0, stream>>>(scores, Vattf, attd);
  k6_xproj<<<48, 256, 0, stream>>>(attd, Wo, bo, outX, sums + 256);
  k7_stats<<<1, 128, 0, stream>>>(sums, stats);
  k8_edge_out<<<36864, 256, 0, stream>>>(e, outE, stats, bew, beb);
  k9_node_out<<<96, 256, 0, stream>>>(x, outX, stats, bnw, bnb);
}
